// ImprovedSGCNModel_77601469104427
// MI455X (gfx1250) — matrix operand order repaired and checked by reading, not compiled, not run on hardware
//
#include <hip/hip_runtime.h>
#include <hip/hip_bf16.h>

// ---------------------------------------------------------------------------
// Types for WMMA (CDNA5 / gfx1250, wave32)
// ---------------------------------------------------------------------------
typedef __attribute__((ext_vector_type(16))) _Float16 v16h;
typedef __attribute__((ext_vector_type(8)))  _Float16 v8h;
typedef __attribute__((ext_vector_type(8)))  float    v8f;

#define HDIM   256
#define NB     8
#define NN     256
#define MROWS  (NB * NN)      // 2048
#define BNEPS  1e-5f

// ---------------------------------------------------------------------------
// A-fragment loader: 16x32 f16 A fragment in the contiguous k order this code uses:
//   lanes 0-15 : row M=lane,    K 0..15 (contiguous, matches B)
//   lanes16-31 : row M=lane-16, K 16..31
// arow points at row start (+k). hs = lane>>4.
// ---------------------------------------------------------------------------
__device__ __forceinline__ v16h load_afrag(const _Float16* arow, int hs) {
    v8h lo = *(const v8h*)(arow + 16 * hs);
    v8h hi = *(const v8h*)(arow + 8 + 16 * hs);
    v16h a;
#pragma unroll
    for (int e = 0; e < 8; ++e) { a[e] = lo[e]; a[e + 8] = hi[e]; }
    return a;
}

#define WMMA_F16(a, b, c) \
    __builtin_amdgcn_wmma_f32_16x16x32_f16(false, (a), false, (b), (short)0, (c), false, false)

// ---------------------------------------------------------------------------
// Generic f32 -> f16 sub-matrix conversion
// ---------------------------------------------------------------------------
__global__ void k_cvt_f16(const float* __restrict__ src, _Float16* __restrict__ dst,
                          int rows, int ld, int coloff, int ncols) {
    int idx = blockIdx.x * blockDim.x + threadIdx.x;
    if (idx >= rows * ncols) return;
    int r = idx / ncols, c = idx - r * ncols;
    dst[idx] = (_Float16)src[(size_t)r * ld + coloff + c];
}

__global__ void k_mask_f16(const int* __restrict__ adj, _Float16* __restrict__ m16) {
    int idx = blockIdx.x * blockDim.x + threadIdx.x;
    if (idx >= NN * NN) return;
    m16[idx] = (_Float16)((adj[idx] > 0) ? 1.0f : 0.0f);
}

// er_sum[b,j,k] = sum_i mask[j,i] * ef[b,i,j,12+k];  deg[j] = sum_i mask[j,i]
__global__ void k_prep(const float* __restrict__ ef, const int* __restrict__ adj,
                       float* __restrict__ er_sum, float* __restrict__ deg) {
    int tid = blockIdx.x * blockDim.x + threadIdx.x;   // 0..2047
    if (tid >= MROWS) return;
    int b = tid >> 8, j = tid & 255;
    float s0 = 0.f, s1 = 0.f, s2 = 0.f, d = 0.f;
    for (int i = 0; i < NN; ++i) {
        float m = (adj[j * NN + i] > 0) ? 1.0f : 0.0f;
        size_t e = (((size_t)(b * NN + i)) * NN + j) * 15 + 12;
        s0 += m * ef[e + 0];
        s1 += m * ef[e + 1];
        s2 += m * ef[e + 2];
        d  += m;
    }
    er_sum[(size_t)tid * 3 + 0] = s0;
    er_sum[(size_t)tid * 3 + 1] = s1;
    er_sum[(size_t)tid * 3 + 2] = s2;
    if (b == 0) deg[j] = d;
}

// raw[row,c] = sum_k nf[row,k]*emb_W[c,k] + emb_b[c]
__global__ void k_emb_raw(const float* __restrict__ nf, const float* __restrict__ W,
                          const float* __restrict__ bias, float* __restrict__ raw) {
    int idx = blockIdx.x * blockDim.x + threadIdx.x;
    if (idx >= MROWS * HDIM) return;
    int row = idx >> 8, c = idx & 255;
    float a = bias[c];
#pragma unroll
    for (int k = 0; k < 6; ++k) a += nf[row * 6 + k] * W[c * 6 + k];
    raw[idx] = a;
}

__global__ void k_zero(float* p, int n) {
    int i = blockIdx.x * blockDim.x + threadIdx.x;
    if (i < n) p[i] = 0.f;
}

// Batchnorm pass 1: per-channel sum / sumsq via float atomics.
__global__ void k_bn_reduce(const float* __restrict__ x, float* __restrict__ sums,
                            float* __restrict__ sumsq) {
    int c = threadIdx.x;
    int r0 = blockIdx.x * 8;
    float s = 0.f, q = 0.f;
#pragma unroll
    for (int r = 0; r < 8; ++r) {
        float v = x[(size_t)(r0 + r) * HDIM + c];
        s += v; q += v * v;
    }
    atomicAdd(&sums[c], s);
    atomicAdd(&sumsq[c], q);
}

// Batchnorm pass 2: normalize (+optional relu, +optional residual), write f32+f16
__global__ void k_bn_apply(const float* __restrict__ x, const float* __restrict__ sums,
                           const float* __restrict__ sumsq, const float* __restrict__ g,
                           const float* __restrict__ bb, const float* __restrict__ residual,
                           int relu, float* __restrict__ out32, _Float16* __restrict__ out16) {
    int idx = blockIdx.x * blockDim.x + threadIdx.x;
    if (idx >= MROWS * HDIM) return;
    int c = idx & 255;
    float mean = sums[c] * (1.0f / MROWS);
    float var  = sumsq[c] * (1.0f / MROWS) - mean * mean;
    float rstd = rsqrtf(var + BNEPS);
    float v = (x[idx] - mean) * rstd * g[c] + bb[c];
    if (relu) v = fmaxf(v, 0.f);
    if (residual) v += residual[idx];
    out32[idx] = v;
    out16[idx] = (_Float16)v;
}

// ---------------------------------------------------------------------------
// WMMA GEMM: C = A @ W^T (+bias, +relu). A:[M,K] f16 row-major, W:[N,K] f16
// row-major (so W^T columns are contiguous -> b128 B-fragment loads).
// Each wave computes a 16x64 tile (4 accumulators share one A fragment).
// Optional batched via grid.z with element strides. Optional f16 shadow of C,
// optionally stored TRANSPOSED per 256-row batch: C16[(row/256)*N+col][row%256]
// (feeds the next GEMM's column-contiguous B operand).
// Block = 128 threads = 4 waves along M; grid = (N/64, M/64, batches).
// ---------------------------------------------------------------------------
__global__ void k_gemm_wt(const _Float16* __restrict__ A, const _Float16* __restrict__ W,
                          float* __restrict__ C, _Float16* __restrict__ C16,
                          const float* __restrict__ bias, int relu, int c16t,
                          int M, int N, int K,
                          long long sA, long long sW, long long sC) {
    int bz = blockIdx.z;
    A += (size_t)bz * sA;
    W += (size_t)bz * sW;
    if (C) C += (size_t)bz * sC;
    int wave = threadIdx.x >> 5, lane = threadIdx.x & 31;
    int lr = lane & 15, hs = lane >> 4;
    int n0 = blockIdx.x * 64;
    int m0 = (blockIdx.y * 4 + wave) * 16;
    if (m0 >= M) return;                       // wave-uniform
    const _Float16* arow = A + (size_t)(m0 + lr) * K;
    const _Float16* bcol = W + (size_t)(n0 + lr) * K + 16 * hs;
    v8f acc0 = {}, acc1 = {}, acc2 = {}, acc3 = {};
#pragma unroll 2
    for (int k = 0; k < K; k += 32) {
        // prefetch next K panel (lowers to global_prefetch_b8; speculative-safe)
        __builtin_prefetch(arow + k + 256, 0, 3);
        __builtin_prefetch(bcol + k + 256, 0, 3);
        v16h a = load_afrag(arow + k, hs);
        v16h b0 = *(const v16h*)(bcol + k);
        v16h b1 = *(const v16h*)(bcol + 16 * (size_t)K + k);
        v16h b2 = *(const v16h*)(bcol + 32 * (size_t)K + k);
        v16h b3 = *(const v16h*)(bcol + 48 * (size_t)K + k);
        acc0 = WMMA_F16(a, b0, acc0);
        acc1 = WMMA_F16(a, b1, acc1);
        acc2 = WMMA_F16(a, b2, acc2);
        acc3 = WMMA_F16(a, b3, acc3);
    }
    v8f accs[4] = {acc0, acc1, acc2, acc3};
#pragma unroll
    for (int nt = 0; nt < 4; ++nt) {
#pragma unroll
        for (int r = 0; r < 8; ++r) {
            int row = m0 + r + 8 * hs, col = n0 + nt * 16 + lr;
            float v = accs[nt][r] + (bias ? bias[col] : 0.f);
            if (relu) v = fmaxf(v, 0.f);
            if (C) C[(size_t)row * N + col] = v;
            if (C16) {
                if (c16t)
                    C16[((size_t)(row >> 8) * N + col) * 256 + (row & 255)] = (_Float16)v;
                else
                    C16[(size_t)row * N + col] = (_Float16)v;
            }
        }
    }
}

// messages -> concat buffer: cat[r, 0..255]=emb16, cat[r, 256+h]=message
__global__ void k_combine(const float* __restrict__ m1, const _Float16* __restrict__ emb16,
                          const float* __restrict__ er_sum, const float* __restrict__ deg,
                          const float* __restrict__ msgW, const float* __restrict__ msgb,
                          int layer, _Float16* __restrict__ cat) {
    int idx = blockIdx.x * blockDim.x + threadIdx.x;
    if (idx >= MROWS * HDIM) return;
    int r = idx >> 8, h = idx & 255;
    int j = r & 255;
    const float* Wrow = msgW + ((size_t)(layer * HDIM + h)) * 259;   // msg_W[l,h,:]
    float v = m1[idx];
#pragma unroll
    for (int k = 0; k < 3; ++k) v += er_sum[(size_t)r * 3 + k] * Wrow[256 + k];
    v += deg[j] * msgb[layer * HDIM + h];
    cat[(size_t)r * 512 + h]       = emb16[idx];
    cat[(size_t)r * 512 + 256 + h] = (_Float16)v;
}

// ---------------------------------------------------------------------------
// Fused pairwise flow head. One block = 64 threads (2 waves) = 32 (i,j) pairs.
//   x = Pi[b,i] + Pj[b,j] + er@W1e^T + t@W1t^T + b1      (128)
//   x = relu(LN(x));  y = x @ W2^T + b2 (WMMA, K=128,N=64)
//   y = relu(LN(y));  flow = relu(y . w3 + b3)
// grid = (N/32, N, B)
// ---------------------------------------------------------------------------
__global__ void __launch_bounds__(64)
k_pairwise(const float* __restrict__ ef, const float* __restrict__ tf,
           const float* __restrict__ Pi, const float* __restrict__ Pj,
           const float* __restrict__ fpW1, const float* __restrict__ b1,
           const float* __restrict__ ln1g, const float* __restrict__ ln1b,
           const _Float16* __restrict__ W2h, const float* __restrict__ b2,
           const float* __restrict__ ln2g, const float* __restrict__ ln2b,
           const float* __restrict__ w3, const float* __restrict__ b3,
           float* __restrict__ out) {
    __shared__ __align__(16) _Float16 w2s[64 * 128];   // 16 KB, col-contig per out row
    __shared__ __align__(16) _Float16 xh[32][136];     // padded: kills A-frag conflicts
    __shared__ float xs[32][129];                      // padded f32 layer-1 acts
    __shared__ float ys[32][65];                       // padded layer-2 acts
    __shared__ float e7[32][8];
    __shared__ float w3s[64];

    int tid = threadIdx.x;                 // 0..63
    int b = blockIdx.z, i = blockIdx.y, j0 = blockIdx.x * 32;
    size_t base_e = ((size_t)(b * NN + i)) * NN + j0;

    { // stage W2 into LDS (vectorized)
        const uint4* s = (const uint4*)W2h;
        uint4* d = (uint4*)w2s;
        for (int t = tid; t < (64 * 128) / 8; t += 64) d[t] = s[t];
    }
    if (tid < 32) {
        size_t e = base_e + tid;
#pragma unroll
        for (int k = 0; k < 3; ++k) e7[tid][k]     = ef[e * 15 + 12 + k];
#pragma unroll
        for (int k = 0; k < 4; ++k) e7[tid][3 + k] = tf[e * 4 + k];
    }
    w3s[tid] = w3[tid & 63];
    __syncthreads();

    // ---- stage 1: build x[32][128]; thread handles channels tid and tid+64
    {
        size_t pib = ((size_t)(b * NN + i)) * 128;
        int c0 = tid, c1 = tid + 64;
        float picA = Pi[pib + c0] + b1[c0];
        float picB = Pi[pib + c1] + b1[c1];
        float wA[7], wB[7];
#pragma unroll
        for (int k = 0; k < 7; ++k) {
            wA[k] = fpW1[(size_t)c0 * 519 + 512 + k];
            wB[k] = fpW1[(size_t)c1 * 519 + 512 + k];
        }
        for (int p = 0; p < 32; ++p) {
            size_t pjb = ((size_t)(b * NN) + j0 + p) * 128;
            float dA = 0.f, dB = 0.f;
#pragma unroll
            for (int k = 0; k < 7; ++k) { dA += e7[p][k] * wA[k]; dB += e7[p][k] * wB[k]; }
            xs[p][c0] = picA + Pj[pjb + c0] + dA;
            xs[p][c1] = picB + Pj[pjb + c1] + dB;
        }
    }
    __syncthreads();

    // ---- layernorm(128) + relu -> f16
    if (tid < 32) {
        float s = 0.f, q = 0.f;
        for (int c = 0; c < 128; ++c) { float v = xs[tid][c]; s += v; q += v * v; }
        float mean = s * (1.0f / 128.0f);
        float var  = q * (1.0f / 128.0f) - mean * mean;
        float rstd = rsqrtf(var + BNEPS);
        for (int c = 0; c < 128; ++c) {
            float v = (xs[tid][c] - mean) * rstd * ln1g[c] + ln1b[c];
            xh[tid][c] = (_Float16)fmaxf(v, 0.f);
        }
    }
    __syncthreads();

    // ---- WMMA: y[32][64] = x @ W2^T ; wave w owns rows [16w, 16w+16)
    {
        int wave = tid >> 5, lane = tid & 31;
        int lr = lane & 15, hs = lane >> 4;
        int m0 = wave * 16;
#pragma unroll
        for (int nt = 0; nt < 4; ++nt) {
            v8f acc = {};
#pragma unroll
            for (int kt = 0; kt < 4; ++kt) {
                v16h a = load_afrag(&xh[m0 + lr][kt * 32], hs);
                v16h bfrag = *(const v16h*)&w2s[(nt * 16 + lr) * 128 + kt * 32 + 16 * hs];
                acc = WMMA_F16(a, bfrag, acc);
            }
#pragma unroll
            for (int r = 0; r < 8; ++r) {
                int row = m0 + r + 8 * hs, col = nt * 16 + lr;
                ys[row][col] = acc[r] + b2[col];
            }
        }
    }
    __syncthreads();

    // ---- layernorm(64) + relu + dot(w3) + relu
    if (tid < 32) {
        float s = 0.f, q = 0.f;
        for (int c = 0; c < 64; ++c) { float v = ys[tid][c]; s += v; q += v * v; }
        float mean = s * (1.0f / 64.0f);
        float var  = q * (1.0f / 64.0f) - mean * mean;
        float rstd = rsqrtf(var + BNEPS);
        float acc = 0.f;
        for (int c = 0; c < 64; ++c) {
            float v = (ys[tid][c] - mean) * rstd * ln2g[c] + ln2b[c];
            acc += fmaxf(v, 0.f) * w3s[c];
        }
        out[base_e + tid] = fmaxf(acc + b3[0], 0.f);
    }
}

// ---------------------------------------------------------------------------
// Host-side launch
// ---------------------------------------------------------------------------
extern "C" void kernel_launch(void* const* d_in, const int* in_sizes, int n_in,
                              void* d_out, int out_size, void* d_ws, size_t ws_size,
                              hipStream_t stream) {
    const float* nf   = (const float*)d_in[0];
    const float* ef   = (const float*)d_in[1];
    const float* tf   = (const float*)d_in[2];
    const int*   adj  = (const int*)d_in[3];
    const float* embW = (const float*)d_in[4];
    const float* embB = (const float*)d_in[5];
    const float* nbg  = (const float*)d_in[6];
    const float* nbb  = (const float*)d_in[7];
    const float* msgW = (const float*)d_in[8];
    const float* msgb = (const float*)d_in[9];
    const float* updW = (const float*)d_in[10];
    const float* updb = (const float*)d_in[11];
    const float* bng  = (const float*)d_in[12];
    const float* bnb  = (const float*)d_in[13];
    const float* fpW1 = (const float*)d_in[14];
    const float* fpb1 = (const float*)d_in[15];
    const float* ln1g = (const float*)d_in[16];
    const float* ln1b = (const float*)d_in[17];
    const float* fpW2 = (const float*)d_in[18];
    const float* fpb2 = (const float*)d_in[19];
    const float* ln2g = (const float*)d_in[20];
    const float* ln2b = (const float*)d_in[21];
    const float* fpW3 = (const float*)d_in[22];
    const float* fpb3 = (const float*)d_in[23];
    float* out = (float*)d_out;

    // ---- workspace carving (~14 MB total)
    char* p = (char*)d_ws;
    auto take = [&](size_t bytes) { char* q = p; p += (bytes + 255) & ~size_t(255); return q; };
    float*    emb32  = (float*)take((size_t)MROWS * HDIM * 4);
    _Float16* emb16  = (_Float16*)take((size_t)MROWS * HDIM * 2);
    float*    raw    = (float*)take((size_t)MROWS * HDIM * 4);
    _Float16* X16T   = (_Float16*)take((size_t)MROWS * HDIM * 2);  // per-batch transposed
    float*    m1     = (float*)take((size_t)MROWS * HDIM * 4);
    _Float16* cat16  = (_Float16*)take((size_t)MROWS * 512 * 2);
    float*    ersum  = (float*)take((size_t)MROWS * 3 * 4);
    float*    deg    = (float*)take(NN * 4);
    float*    sums   = (float*)take(HDIM * 4);
    float*    sumsq  = (float*)take(HDIM * 4);
    _Float16* wn16   = (_Float16*)take((size_t)3 * HDIM * HDIM * 2);
    _Float16* uw16   = (_Float16*)take((size_t)3 * HDIM * 512 * 2);
    _Float16* w1i16  = (_Float16*)take((size_t)128 * HDIM * 2);
    _Float16* w1j16  = (_Float16*)take((size_t)128 * HDIM * 2);
    _Float16* w2h16  = (_Float16*)take((size_t)64 * 128 * 2);
    _Float16* mask16 = (_Float16*)take((size_t)NN * NN * 2);
    float*    Pi     = (float*)take((size_t)MROWS * 128 * 4);
    float*    Pj     = (float*)take((size_t)MROWS * 128 * 4);
    (void)ws_size; (void)n_in; (void)in_sizes; (void)out_size;

    const int T = 256;
    auto blks = [&](long long n) { return (unsigned)((n + T - 1) / T); };

    // ---- weight conversions (once per call)
    k_cvt_f16<<<blks(768LL * 256), T, 0, stream>>>(msgW, wn16, 768, 259, 0, 256);
    k_cvt_f16<<<blks(768LL * 512), T, 0, stream>>>(updW, uw16, 768, 512, 0, 512);
    k_cvt_f16<<<blks(128LL * 256), T, 0, stream>>>(fpW1, w1i16, 128, 519, 0, 256);
    k_cvt_f16<<<blks(128LL * 256), T, 0, stream>>>(fpW1, w1j16, 128, 519, 256, 256);
    k_cvt_f16<<<blks(64LL * 128),  T, 0, stream>>>(fpW2, w2h16, 64, 128, 0, 128);
    k_mask_f16<<<blks(NN * NN), T, 0, stream>>>(adj, mask16);
    k_prep<<<blks(MROWS), T, 0, stream>>>(ef, adj, ersum, deg);

    // ---- embedding + batchnorm + relu
    k_emb_raw<<<blks((long long)MROWS * HDIM), T, 0, stream>>>(nf, embW, embB, raw);
    k_zero<<<2, T, 0, stream>>>(sums, 2 * HDIM);   // sums & sumsq are contiguous
    k_bn_reduce<<<MROWS / 8, HDIM, 0, stream>>>(raw, sums, sumsq);
    k_bn_apply<<<blks((long long)MROWS * HDIM), T, 0, stream>>>(
        raw, sums, sumsq, nbg, nbb, nullptr, 1, emb32, emb16);

    // ---- message passing layers
    for (int l = 0; l < 3; ++l) {
        // X = emb @ Wn^T ; store only a per-batch TRANSPOSED f16 shadow (X16T)
        k_gemm_wt<<<dim3(HDIM / 64, MROWS / 64, 1), 128, 0, stream>>>(
            emb16, wn16 + (size_t)l * HDIM * HDIM, nullptr, X16T, nullptr, 0, 1,
            MROWS, HDIM, HDIM, 0, 0, 0);
        // m1[b] = mask @ X[b] : X16T columns are contiguous -> col-contig B path
        k_gemm_wt<<<dim3(HDIM / 64, NN / 64, NB), 128, 0, stream>>>(
            mask16, X16T, m1, nullptr, nullptr, 0, 0,
            NN, HDIM, NN, 0LL, (long long)NN * HDIM, (long long)NN * HDIM);
        // messages + concat
        k_combine<<<blks((long long)MROWS * HDIM), T, 0, stream>>>(
            m1, emb16, ersum, deg, msgW, msgb, l, cat16);
        // updated = relu(cat @ Wu^T + b)  (K=512)
        k_gemm_wt<<<dim3(HDIM / 64, MROWS / 64, 1), 128, 0, stream>>>(
            cat16, uw16 + (size_t)l * HDIM * 512, raw, nullptr,
            updb + l * HDIM, 1, 0, MROWS, HDIM, 512, 0, 0, 0);
        // batchnorm + residual
        k_zero<<<2, T, 0, stream>>>(sums, 2 * HDIM);
        k_bn_reduce<<<MROWS / 8, HDIM, 0, stream>>>(raw, sums, sumsq);
        k_bn_apply<<<blks((long long)MROWS * HDIM), T, 0, stream>>>(
            raw, sums, sumsq, bng + l * HDIM, bnb + l * HDIM, emb32, 0, emb32, emb16);
    }

    // ---- pairwise head projections: Pi = emb@W1i^T, Pj = emb@W1j^T
    k_gemm_wt<<<dim3(128 / 64, MROWS / 64, 1), 128, 0, stream>>>(
        emb16, w1i16, Pi, nullptr, nullptr, 0, 0, MROWS, 128, HDIM, 0, 0, 0);
    k_gemm_wt<<<dim3(128 / 64, MROWS / 64, 1), 128, 0, stream>>>(
        emb16, w1j16, Pj, nullptr, nullptr, 0, 0, MROWS, 128, HDIM, 0, 0, 0);

    // ---- fused pairwise MLP (dominant compute, WMMA in LDS)
    k_pairwise<<<dim3(NN / 32, NN, NB), 64, 0, stream>>>(
        ef, tf, Pi, Pj, fpW1, fpb1, ln1g, ln1b, w2h16, fpb2, ln2g, ln2b,
        fpW3, fpb3, out);
}
